// BertSelfAttention_36103495090557
// MI455X (gfx1250) — matrix operand order repaired and checked by reading, not compiled, not run on hardware
//
#include <hip/hip_runtime.h>
#include <cstdint>
#include <cstddef>

// BERT self-attention for MI455X (gfx1250), wave32 + WMMA bf16.
// Pipeline: f32->bf16 convert; QKV GEMMs (v_wmma_f32_16x16x32_bf16) with
// double-buffered LDS fed by GLOBAL_LOAD_ASYNC_TO_LDS (A tiles) and the
// Tensor Data Mover (B/weight tiles, with TDM padding to keep an 80B LDS row
// pitch = bank-conflict-free); flash-attention per (b,h,128-q-rows) with
// async-staged K tiles; output GEMM to f32.
// Workspace assumption: ws_size >= ~68 MB. Deterministic, all buffers written
// before read.

typedef __bf16 bf16_t;
typedef __attribute__((ext_vector_type(16))) __bf16 v16bf;
typedef __attribute__((ext_vector_type(8)))  float  v8f;
typedef __attribute__((ext_vector_type(4)))  unsigned v4u;
typedef __attribute__((ext_vector_type(8)))  unsigned v8u;

#define NBATCH 8
#define SEQ    1024
#define HID    768
#define NHEAD  12
#define HDIM   64
#define MTOT   (NBATCH * SEQ)   // 8192 rows

union Frag  { v16bf v; unsigned u[8]; };
union Acc   { v8f   v; float    f[8]; };
union Pack16 { uint4 q[2]; bf16_t h[16]; };

// ---------------------------------------------------------------------------
// CDNA5 async / TDM data movers (inline asm; ASYNCcnt + TENSORcnt tracked)
// ---------------------------------------------------------------------------
__device__ __forceinline__ void async_load_b128(unsigned lds_byte_off,
                                                const void* gbase,
                                                unsigned gbyte_off) {
  // GVS mode: mem = SGPR64 base + VGPR32 offset; LDS address from the VDST VGPR.
  asm volatile("global_load_async_to_lds_b128 %0, %1, %2"
               :: "v"(lds_byte_off), "v"(gbyte_off), "s"(gbase)
               : "memory");
}
__device__ __forceinline__ void wait_async0() {
  asm volatile("s_wait_asynccnt 0x0" ::: "memory");
}

// TDM: 2D tile (tile_rows x tile_k bf16 elems) from row-major global
// (row pitch = row_stride elems) into LDS. pad_enable inserts 16B after every
// 64B so LDS row pitch is 80B (= 40 bf16 = our LDS stride).
__device__ __forceinline__ void tdm_load_tile_2d(unsigned lds_byte_off,
                                                 const void* gptr,
                                                 unsigned tile_k,
                                                 unsigned tile_rows,
                                                 unsigned row_stride,
                                                 unsigned tensor_d0,
                                                 unsigned tensor_d1) {
  const unsigned long long ga = (unsigned long long)(uintptr_t)gptr;
  v4u g0;
  g0.x = 1u;                                   // count=1, user descriptor
  g0.y = lds_byte_off;                         // lds_addr
  g0.z = (unsigned)(ga & 0xffffffffu);         // global_addr[31:0]
  g0.w = (unsigned)((ga >> 32) & 0x01ffffffu)  // global_addr[56:32]
       | (2u << 30);                           // type = 2 ("image")
  v8u g1;
  g1.s0 = (1u << 16)                           // data_size = 1 -> 2 bytes
        | (1u << 20)                           // pad_enable
        | (3u << 22)                           // pad_interval: 16 dwords (64B)
        | (3u << 25);                          // pad_amount: 4 dwords (16B)
  g1.s1 = (tensor_d0 & 0xffffu) << 16;                       // tensor_dim0[15:0]
  g1.s2 = ((tensor_d0 >> 16) & 0xffffu)
        | ((tensor_d1 & 0xffffu) << 16);                     // dim0[31:16] | dim1[15:0]
  g1.s3 = ((tensor_d1 >> 16) & 0xffffu)
        | ((tile_k & 0xffffu) << 16);                        // dim1[31:16] | tile_dim0
  g1.s4 = (tile_rows & 0xffffu);                             // tile_dim1 (tile_dim2=0)
  g1.s5 = row_stride;                                        // tensor_dim0_stride[31:0]
  g1.s6 = 0u;
  g1.s7 = 0u;
  asm volatile("tensor_load_to_lds %0, %1" :: "s"(g0), "s"(g1) : "memory");
}

// ---------------------------------------------------------------------------
// WMMA fragment loaders (layouts per CDNA5 ISA §7.12.2, wave32)
// A 16x32 bf16: lane -> row = lane%16; VGPR v -> k = (v/4)*16 + (lane/16)*8 + (v%4)*2
// B 32x16 bf16: lane -> col = lane%16; VGPR v -> k = (v/4)*16 + (lane/16)*8 + (v%4)*2
// C 16x16 f32 : lane -> col = lane%16; slot i -> row = i + 8*(lane/16)
// ---------------------------------------------------------------------------
__device__ __forceinline__ void load_a_frag(Frag& f, const bf16_t* base, int stride) {
  const int lane = threadIdx.x & 31;
  const int row  = lane & 15;
  const int half = lane >> 4;
  const bf16_t* p = base + row * stride;
#pragma unroll
  for (int v = 0; v < 8; ++v) {
    const int k = ((v >> 2) << 4) + (half << 3) + ((v & 3) << 1);
    f.u[v] = *reinterpret_cast<const unsigned*>(p + k);
  }
}

__device__ __forceinline__ void load_b_frag(Frag& f, const bf16_t* base, int stride) {
  const int lane = threadIdx.x & 31;
  const int n    = lane & 15;
  const int half = lane >> 4;
  const bf16_t* p = base + n * stride;
#pragma unroll
  for (int v = 0; v < 8; ++v)
    f.u[v] = *reinterpret_cast<const unsigned*>(p + ((v >> 2) << 4) + (half << 3) + ((v & 3) << 1));
}

__device__ __forceinline__ v8f wmma_bf16(const Frag& a, const Frag& b, v8f c) {
  return __builtin_amdgcn_wmma_f32_16x16x32_bf16(false, a.v, false, b.v,
                                                 (short)0, c, false, false);
}

// ---------------------------------------------------------------------------
// Stage 0: f32 -> bf16 conversion (n multiple of 4)
// ---------------------------------------------------------------------------
__global__ void cvt_f32_bf16(const float* __restrict__ src,
                             bf16_t* __restrict__ dst, int n4) {
  int i = blockIdx.x * blockDim.x + threadIdx.x;
  if (i < n4) {
    float4 f = reinterpret_cast<const float4*>(src)[i];
    dst[4 * i + 0] = (bf16_t)f.x;
    dst[4 * i + 1] = (bf16_t)f.y;
    dst[4 * i + 2] = (bf16_t)f.z;
    dst[4 * i + 3] = (bf16_t)f.w;
  }
}

// ---------------------------------------------------------------------------
// Stage 1/3: C = (A @ W^T + bias) * scale, A [MTOT,K] bf16, W [N,K] bf16.
// Block tile 128(M) x 64(N), 8 waves (4 M x 2 N), wave tile 32x32.
// Double-buffered LDS: A tile via async-to-LDS, B tile via TDM (padded pitch).
//   mode 0: bf16 out scattered to [B, NH, L, HD]   (QKV projections)
//   mode 1: f32  out [MTOT, N]                     (final projection)
// ---------------------------------------------------------------------------
__global__ __launch_bounds__(256) void gemm_bf16_kernel(
    const bf16_t* __restrict__ A, const bf16_t* __restrict__ W,
    const float* __restrict__ bias, float scale, int mode,
    bf16_t* __restrict__ outb, float* __restrict__ outf, int K, int N) {
  constexpr int LDA = 40;  // 80B pitch: 16B-aligned, conflict-free column reads
  constexpr int LDB = 40;  // matches TDM pad settings (64B data + 16B pad)
  __shared__ bf16_t As[2][128 * LDA];
  __shared__ bf16_t Bs[2][64 * LDB];

  const int tid  = threadIdx.x;
  const int lane = tid & 31;
  const int wave = tid >> 5;
  const int wm   = wave & 3;
  const int wn   = wave >> 2;
  const int half = lane >> 4;
  const int col  = lane & 15;
  const int m0   = blockIdx.y * 128;
  const int n0   = blockIdx.x * 64;

  Acc acc[2][2];
#pragma unroll
  for (int i = 0; i < 2; ++i)
#pragma unroll
    for (int j = 0; j < 2; ++j)
      acc[i][j].v = (v8f){0.f, 0.f, 0.f, 0.f, 0.f, 0.f, 0.f, 0.f};

  const int arow = tid >> 1, ahalf = tid & 1;  // A: 128 rows x 2 x 16 bf16

  auto stage = [&](int buf, int kt) {
    // A tile: per-lane async 2x16B into LDS (ASYNCcnt).
    const unsigned lds0 =
        (unsigned)(uintptr_t)&As[buf][arow * LDA + ahalf * 16];
    const unsigned go =
        (unsigned)(((size_t)(m0 + arow) * K + kt + ahalf * 16) * sizeof(bf16_t));
    async_load_b128(lds0, A, go);
    async_load_b128(lds0 + 16, A, go + 16);
    // B tile: one TDM descriptor per block (wave 0 only; TENSORcnt).
    if (wave == 0) {
      tdm_load_tile_2d((unsigned)(uintptr_t)&Bs[buf][0],
                       W + (size_t)n0 * K + kt,
                       /*tile_k=*/32, /*tile_rows=*/64,
                       /*row_stride=*/(unsigned)K,
                       /*tensor_d0=*/(unsigned)K, /*tensor_d1=*/64u);
    }
  };

  stage(0, 0);
  wait_async0();
  if (wave == 0) __builtin_amdgcn_s_wait_tensorcnt(0);
  __syncthreads();

  int buf = 0;
  for (int kt = 0; kt < K; kt += 32) {
    if (kt + 32 < K) stage(buf ^ 1, kt + 32);   // DMA next tile under compute

    Frag a0, a1, b0, b1;
    load_a_frag(a0, &As[buf][(wm * 32 + 0)  * LDA], LDA);
    load_a_frag(a1, &As[buf][(wm * 32 + 16) * LDA], LDA);
    load_b_frag(b0, &Bs[buf][(wn * 32 + 0)  * LDB], LDB);
    load_b_frag(b1, &Bs[buf][(wn * 32 + 16) * LDB], LDB);

    acc[0][0].v = wmma_bf16(a0, b0, acc[0][0].v);
    acc[0][1].v = wmma_bf16(a0, b1, acc[0][1].v);
    acc[1][0].v = wmma_bf16(a1, b0, acc[1][0].v);
    acc[1][1].v = wmma_bf16(a1, b1, acc[1][1].v);

    wait_async0();
    if (wave == 0) __builtin_amdgcn_s_wait_tensorcnt(0);
    __syncthreads();
    buf ^= 1;
  }

#pragma unroll
  for (int i = 0; i < 2; ++i)
#pragma unroll
    for (int j = 0; j < 2; ++j) {
      const int n = n0 + wn * 32 + j * 16 + col;
      const float bn = bias[n];
#pragma unroll
      for (int sl = 0; sl < 8; ++sl) {
        const int m = m0 + wm * 32 + i * 16 + sl + 8 * half;
        const float val = (acc[i][j].f[sl] + bn) * scale;
        if (mode == 0) {
          const int bb = m >> 10, ll = m & (SEQ - 1);
          const int hh = n >> 6,  dd = n & (HDIM - 1);
          outb[(((size_t)bb * NHEAD + hh) * SEQ + ll) * HDIM + dd] = (bf16_t)val;
        } else {
          outf[(size_t)m * N + n] = val;
        }
      }
    }
}

// ---------------------------------------------------------------------------
// Stage 2: flash attention. Q pre-scaled by 1/sqrt(HD).
// Block = 8 waves, 128 q rows of one (b,h); wave owns 16 rows -> softmax is
// wave-local (butterfly over 16-lane halves matches WMMA C layout).
// K/V tiles double-buffered; K staged via async-to-LDS, V transposed manually.
// ---------------------------------------------------------------------------
__global__ __launch_bounds__(256) void attn_kernel(
    const bf16_t* __restrict__ Qg, const bf16_t* __restrict__ Kg,
    const bf16_t* __restrict__ Vg, const float* __restrict__ maskg,
    bf16_t* __restrict__ ctx) {
  constexpr int LDK = 72;  // 144B rows (16B-aligned)
  constexpr int LDP = 66;  // 132B rows (4B-aligned), 64 cols + pad
  __shared__ bf16_t Ks[2][64 * LDK];
  __shared__ bf16_t Vt[2][64 * LDK];
  __shared__ bf16_t Ps[8 * 16 * LDP];

  const int qt = blockIdx.x & 7;                 // L/128 = 8 q-tiles
  const int h  = (blockIdx.x >> 3) % NHEAD;
  const int b  = blockIdx.x / (8 * NHEAD);

  const int tid  = threadIdx.x;
  const int lane = tid & 31;
  const int wave = tid >> 5;
  const int half = lane >> 4;
  const int col  = lane & 15;

  const size_t head_base = ((size_t)b * NHEAD + h) * SEQ;
  const int qbase = qt * 128 + wave * 16;

  Frag aq0, aq1;
  {
    const bf16_t* qp = Qg + (head_base + qbase) * HDIM;
    load_a_frag(aq0, qp, HDIM);        // d = 0..31
    load_a_frag(aq1, qp + 32, HDIM);   // d = 32..63
  }

  bf16_t* Pw = Ps + wave * 16 * LDP;

  float mrow[8], lrow[8];
#pragma unroll
  for (int sl = 0; sl < 8; ++sl) { mrow[sl] = -3.0e38f; lrow[sl] = 0.f; }
  Acc o[4];
#pragma unroll
  for (int j = 0; j < 4; ++j)
    o[j].v = (v8f){0.f, 0.f, 0.f, 0.f, 0.f, 0.f, 0.f, 0.f};

  const int krow = tid >> 2, kc = tid & 3;     // K coop: 64 rows x 4 x 8 bf16
  const int vkey = tid >> 2, vpart = tid & 3;  // V coop: 64 keys x 4 x 16 d

  auto stageK = [&](int buf, int jt) {
    const unsigned lds = (unsigned)(uintptr_t)&Ks[buf][krow * LDK + kc * 8];
    const unsigned go =
        (unsigned)(((head_base + jt * 64 + krow) * HDIM + kc * 8) * sizeof(bf16_t));
    async_load_b128(lds, Kg, go);
  };
  auto stageV = [&](int buf, int jt) {
    const uint4* vp = reinterpret_cast<const uint4*>(
        Vg + (head_base + jt * 64 + vkey) * HDIM + vpart * 16);
    Pack16 pk;
    pk.q[0] = vp[0];
    pk.q[1] = vp[1];
#pragma unroll
    for (int e = 0; e < 16; ++e)
      Vt[buf][(vpart * 16 + e) * LDK + vkey] = pk.h[e];
  };

  stageK(0, 0);
  stageV(0, 0);
  wait_async0();
  __syncthreads();

  int buf = 0;
  for (int jt = 0; jt < SEQ / 64; ++jt) {
    if (jt + 1 < SEQ / 64) {                    // DMA next tile under compute
      stageK(buf ^ 1, jt + 1);
      stageV(buf ^ 1, jt + 1);
    }

    // --- S = Q K^T (+mask): 4 n-subtiles of 16 keys, K-dim = 64 (2 steps) ---
    Acc s[4];
#pragma unroll
    for (int j = 0; j < 4; ++j) {
      s[j].v = (v8f){0.f, 0.f, 0.f, 0.f, 0.f, 0.f, 0.f, 0.f};
      Frag kb;
      load_b_frag(kb, &Ks[buf][(j * 16) * LDK], LDK);        // d 0..31
      s[j].v = wmma_bf16(aq0, kb, s[j].v);
      load_b_frag(kb, &Ks[buf][(j * 16) * LDK + 32], LDK);   // d 32..63
      s[j].v = wmma_bf16(aq1, kb, s[j].v);
      const float mv = maskg[(size_t)b * SEQ + jt * 64 + j * 16 + col];
#pragma unroll
      for (int sl = 0; sl < 8; ++sl) s[j].f[sl] += mv;
    }

    // --- online softmax (row = slot + 8*half) ---
    float alpha[8];
#pragma unroll
    for (int sl = 0; sl < 8; ++sl) {
      float v = fmaxf(fmaxf(s[0].f[sl], s[1].f[sl]), fmaxf(s[2].f[sl], s[3].f[sl]));
#pragma unroll
      for (int d = 1; d < 16; d <<= 1) v = fmaxf(v, __shfl_xor(v, d, 32));
      const float mn = fmaxf(mrow[sl], v);
      alpha[sl] = __expf(mrow[sl] - mn);
      mrow[sl]  = mn;
    }

    float rs[8];
#pragma unroll
    for (int sl = 0; sl < 8; ++sl) rs[sl] = 0.f;
#pragma unroll
    for (int j = 0; j < 4; ++j)
#pragma unroll
      for (int sl = 0; sl < 8; ++sl) {
        const float p = __expf(s[j].f[sl] - mrow[sl]);
        rs[sl] += p;
        Pw[(sl + 8 * half) * LDP + j * 16 + col] = (bf16_t)p;  // C->A relayout
      }
#pragma unroll
    for (int sl = 0; sl < 8; ++sl) {
      float v = rs[sl];
#pragma unroll
      for (int d = 1; d < 16; d <<= 1) v += __shfl_xor(v, d, 32);
      lrow[sl] = lrow[sl] * alpha[sl] + v;
    }
#pragma unroll
    for (int j = 0; j < 4; ++j)
#pragma unroll
      for (int sl = 0; sl < 8; ++sl) o[j].f[sl] *= alpha[sl];

    // --- O += P V : A = P (16x64 keys), B = Vt (keys x d) ---
    Frag ap0, ap1;
    load_a_frag(ap0, Pw, LDP);        // keys 0..31
    load_a_frag(ap1, Pw + 32, LDP);   // keys 32..63
#pragma unroll
    for (int j = 0; j < 4; ++j) {
      Frag vb;
      load_b_frag(vb, &Vt[buf][(j * 16) * LDK], LDK);        // keys 0..31
      o[j].v = wmma_bf16(ap0, vb, o[j].v);
      load_b_frag(vb, &Vt[buf][(j * 16) * LDK + 32], LDK);   // keys 32..63
      o[j].v = wmma_bf16(ap1, vb, o[j].v);
    }

    wait_async0();
    __syncthreads();
    buf ^= 1;
  }

  // --- normalize and write ctx as [B, L, H] bf16 ---
  float inv[8];
#pragma unroll
  for (int sl = 0; sl < 8; ++sl) inv[sl] = 1.f / fmaxf(lrow[sl], 1e-20f);
#pragma unroll
  for (int j = 0; j < 4; ++j)
#pragma unroll
    for (int sl = 0; sl < 8; ++sl) {
      const int q  = qbase + sl + 8 * half;
      const int dd = j * 16 + col;
      ctx[((size_t)b * SEQ + q) * HID + h * HDIM + dd] = (bf16_t)(o[j].f[sl] * inv[sl]);
    }
}

// ---------------------------------------------------------------------------
extern "C" void kernel_launch(void* const* d_in, const int* in_sizes, int n_in,
                              void* d_out, int out_size, void* d_ws, size_t ws_size,
                              hipStream_t stream) {
  const float* hs   = (const float*)d_in[0];
  const float* mask = (const float*)d_in[1];
  const float* Wq   = (const float*)d_in[2];
  const float* bq   = (const float*)d_in[3];
  const float* Wk   = (const float*)d_in[4];
  const float* bk   = (const float*)d_in[5];
  const float* Wv   = (const float*)d_in[6];
  const float* bv   = (const float*)d_in[7];
  const float* Wo   = (const float*)d_in[8];
  const float* bo   = (const float*)d_in[9];
  (void)in_sizes; (void)n_in; (void)out_size; (void)ws_size;

  char* ws = (char*)d_ws;
  size_t off = 0;
  auto carve = [&](size_t bytes) {
    char* p = ws + off;
    off += (bytes + 255) & ~(size_t)255;
    return p;
  };
  const size_t actB = (size_t)MTOT * HID * sizeof(bf16_t);   // 12.6 MB
  const size_t wgtB = (size_t)HID * HID * sizeof(bf16_t);    // 1.2 MB
  bf16_t* Xb   = (bf16_t*)carve(actB);
  bf16_t* Wqb  = (bf16_t*)carve(wgtB);
  bf16_t* Wkb  = (bf16_t*)carve(wgtB);
  bf16_t* Wvb  = (bf16_t*)carve(wgtB);
  bf16_t* Wob  = (bf16_t*)carve(wgtB);
  bf16_t* Qb   = (bf16_t*)carve(actB);   // [B,NH,L,HD]
  bf16_t* Kb   = (bf16_t*)carve(actB);
  bf16_t* Vb   = (bf16_t*)carve(actB);
  bf16_t* Ctxb = (bf16_t*)carve(actB);   // [B,L,H]

  auto cvt = [&](const float* s, bf16_t* d, size_t n) {
    const int n4 = (int)(n / 4);
    cvt_f32_bf16<<<(n4 + 255) / 256, 256, 0, stream>>>(s, d, n4);
  };
  cvt(hs, Xb, (size_t)MTOT * HID);
  cvt(Wq, Wqb, (size_t)HID * HID);
  cvt(Wk, Wkb, (size_t)HID * HID);
  cvt(Wv, Wvb, (size_t)HID * HID);
  cvt(Wo, Wob, (size_t)HID * HID);

  const dim3 ggrid(HID / 64, MTOT / 128);  // (12, 64)
  gemm_bf16_kernel<<<ggrid, 256, 0, stream>>>(Xb, Wqb, bq, 0.125f, 0, Qb, nullptr, HID, HID);
  gemm_bf16_kernel<<<ggrid, 256, 0, stream>>>(Xb, Wkb, bk, 1.0f,   0, Kb, nullptr, HID, HID);
  gemm_bf16_kernel<<<ggrid, 256, 0, stream>>>(Xb, Wvb, bv, 1.0f,   0, Vb, nullptr, HID, HID);

  attn_kernel<<<NBATCH * NHEAD * (SEQ / 128), 256, 0, stream>>>(Qb, Kb, Vb, mask, Ctxb);

  gemm_bf16_kernel<<<ggrid, 256, 0, stream>>>(Ctxb, Wob, bo, 1.0f, 1, nullptr,
                                              (float*)d_out, HID, HID);
}
